// NegativeSelectionTorch_87780541595761
// MI455X (gfx1250) — matrix operand order repaired and checked by reading, not compiled, not run on hardware
//
#include <hip/hip_runtime.h>
#include <hip/hip_bf16.h>

typedef __attribute__((ext_vector_type(16))) __bf16        v16bf;
typedef __attribute__((ext_vector_type(8)))  float         v8f;
typedef __attribute__((ext_vector_type(8)))  unsigned int  v8u;

#define DIMS   128
#define R_SELF 0.1f

// ---------- fp32 -> bf16 split helpers (RNE via bit math) ----------
__device__ __forceinline__ unsigned short f2bf_rne(float f) {
  unsigned int u = __builtin_bit_cast(unsigned int, f);
  unsigned int r = u + 0x7FFFu + ((u >> 16) & 1u);
  return (unsigned short)(r >> 16);
}
__device__ __forceinline__ float bf2f(unsigned short b) {
  return __builtin_bit_cast(float, (unsigned int)b << 16);
}

// Assemble a v16bf fragment from two 16-byte chunks of a bf16 array.
__device__ __forceinline__ v16bf load_frag2(const unsigned short* p0,
                                            const unsigned short* p1) {
  uint4 a = *reinterpret_cast<const uint4*>(p0);
  uint4 b = *reinterpret_cast<const uint4*>(p1);
  v8u t = {a.x, a.y, a.z, a.w, b.x, b.y, b.z, b.w};
  return __builtin_bit_cast(v16bf, t);
}
__device__ __forceinline__ v16bf load_frag(const unsigned short* p) {
  return load_frag2(p, p + 8);
}

// ---------- Pass 1: split fp32 rows into bf16 hi/lo + exact fp32 row norms ----------
__global__ __launch_bounds__(128)
void split_rows(const float* __restrict__ in,
                unsigned short* __restrict__ hi,
                unsigned short* __restrict__ lo,
                float* __restrict__ norms, int rows) {
  const int row  = blockIdx.x * 4 + (threadIdx.x >> 5);
  if (row >= rows) return;
  const int lane = threadIdx.x & 31;

  const float4 q = *reinterpret_cast<const float4*>(in + (size_t)row * DIMS + lane * 4);
  float fv[4] = {q.x, q.y, q.z, q.w};
  unsigned short h[4], l[4];
  float nsum = 0.f;
  #pragma unroll
  for (int t = 0; t < 4; ++t) {
    float f = fv[t];
    nsum += f * f;                          // norm in full fp32
    h[t] = f2bf_rne(f);
    l[t] = f2bf_rne(f - bf2f(h[t]));
  }
  ushort4 hv = {h[0], h[1], h[2], h[3]};
  ushort4 lv = {l[0], l[1], l[2], l[3]};
  *reinterpret_cast<ushort4*>(hi + (size_t)row * DIMS + lane * 4) = hv;
  *reinterpret_cast<ushort4*>(lo + (size_t)row * DIMS + lane * 4) = lv;

  nsum += __shfl_xor(nsum, 1, 32);
  nsum += __shfl_xor(nsum, 2, 32);
  nsum += __shfl_xor(nsum, 4, 32);
  nsum += __shfl_xor(nsum, 8, 32);
  nsum += __shfl_xor(nsum, 16, 32);
  if (lane == 0) norms[row] = nsum;
}

// ---------- Pass 2: WMMA min-distance ----------
// 8 waves / block over 64 x-rows: waves 0-3 scan columns [0,M/2), waves 4-7 scan
// [M/2,M) over the same rows (doubles resident waves per SIMD for latency hiding).
// Each wave: A fragments pinned in VGPRs; two 16-column s-tiles per iteration ->
// two independent WMMA chains. Partial minima merged through LDS.
__global__ __launch_bounds__(256)
void nsa_mindist_wmma(const unsigned short* __restrict__ xh,
                      const unsigned short* __restrict__ xl,
                      const float* __restrict__ xn,
                      const unsigned short* __restrict__ sh,
                      const unsigned short* __restrict__ sl,
                      const float* __restrict__ sn,
                      float* __restrict__ out, int N, int M) {
  __shared__ float partial[8][16];

  const int lane = threadIdx.x & 31;
  const int wave = threadIdx.x >> 5;       // 0..7
  const int half = lane >> 4;
  const int l15  = lane & 15;

  const int tileInBlock = wave & 3;        // which 16-row tile of this block
  const int mhalf       = wave >> 2;       // which half of the column range
  const int rowBase     = blockIdx.x * 64 + tileInBlock * 16;
  const bool active     = rowBase < N;     // wave-uniform

  float rmin[8];
  #pragma unroll
  for (int p = 0; p < 8; ++p) rmin[p] = 3.0e38f;

  if (active) {                            // wave-uniform: EXEC all-1s inside
    // A fragments (row = lane&15; elems 0..15 = K[16h..+16), matching B)
    // -> two b128 loads per fragment.
    v16bf Ah[4], Al[4];
    const unsigned short* xrh = xh + (size_t)(rowBase + l15) * DIMS;
    const unsigned short* xrl = xl + (size_t)(rowBase + l15) * DIMS;
    #pragma unroll
    for (int kk = 0; kk < 4; ++kk) {
      const int k0 = kk * 32 + 16 * half;
      Ah[kk] = load_frag2(xrh + k0, xrh + k0 + 8);
      Al[kk] = load_frag2(xrl + k0, xrl + k0 + 8);
    }

    const float a2row = xn[rowBase + l15];
    float a2v[8];
    #pragma unroll
    for (int p = 0; p < 8; ++p)
      a2v[p] = __shfl(a2row, p + 8 * half, 32); // C layout: VGPR p, half -> row p+8*half

    // B layout (32x16): lane -> col = lane&15, element e = K[16*half + e] -> one
    // contiguous 32B run per k-step.
    const size_t colOff = (size_t)l15 * DIMS + 16 * half;
    const int jBeg = mhalf ? (M >> 1) : 0;
    const int jEnd = mhalf ? M : (M >> 1);

    for (int j = jBeg; j < jEnd; j += 32) {
      const unsigned short* bh0 = sh + (size_t)j * DIMS + colOff;
      const unsigned short* bl0 = sl + (size_t)j * DIMS + colOff;
      const unsigned short* bh1 = bh0 + 16 * DIMS;
      const unsigned short* bl1 = bl0 + 16 * DIMS;
      if (j + 32 < jEnd) {
        __builtin_prefetch(bh0 + 32 * DIMS, 0, 0);   // global_prefetch_b8
        __builtin_prefetch(bl0 + 32 * DIMS, 0, 0);
        __builtin_prefetch(bh1 + 32 * DIMS, 0, 0);
        __builtin_prefetch(bl1 + 32 * DIMS, 0, 0);
      }
      const float b2_0 = sn[j + l15];                // |s_col|^2, exact fp32
      const float b2_1 = sn[j + 16 + l15];

      v8f c0 = {};
      v8f c1 = {};
      #pragma unroll
      for (int kk = 0; kk < 4; ++kk) {
        v16bf Bh0 = load_frag(bh0 + kk * 32);
        v16bf Bl0 = load_frag(bl0 + kk * 32);
        v16bf Bh1 = load_frag(bh1 + kk * 32);
        v16bf Bl1 = load_frag(bl1 + kk * 32);
        // two independent accumulation chains -> scheduler interleaves them
        c0 = __builtin_amdgcn_wmma_f32_16x16x32_bf16(false, Ah[kk], false, Bh0, (short)0, c0, false, false);
        c1 = __builtin_amdgcn_wmma_f32_16x16x32_bf16(false, Ah[kk], false, Bh1, (short)0, c1, false, false);
        c0 = __builtin_amdgcn_wmma_f32_16x16x32_bf16(false, Ah[kk], false, Bl0, (short)0, c0, false, false);
        c1 = __builtin_amdgcn_wmma_f32_16x16x32_bf16(false, Ah[kk], false, Bl1, (short)0, c1, false, false);
        c0 = __builtin_amdgcn_wmma_f32_16x16x32_bf16(false, Al[kk], false, Bh0, (short)0, c0, false, false);
        c1 = __builtin_amdgcn_wmma_f32_16x16x32_bf16(false, Al[kk], false, Bh1, (short)0, c1, false, false);
      }
      #pragma unroll
      for (int p = 0; p < 8; ++p) {
        float d2a = fmaf(-2.f, c0[p], a2v[p] + b2_0);
        float d2b = fmaf(-2.f, c1[p], a2v[p] + b2_1);
        rmin[p] = fminf(rmin[p], fminf(d2a, d2b));
      }
    }

    // Min across the 16 lanes (columns) of this half.
    #pragma unroll
    for (int p = 0; p < 8; ++p) {
      float v = rmin[p];
      v = fminf(v, __shfl_xor(v, 1, 32));
      v = fminf(v, __shfl_xor(v, 2, 32));
      v = fminf(v, __shfl_xor(v, 4, 32));
      v = fminf(v, __shfl_xor(v, 8, 32));
      rmin[p] = v;
    }
  }

  // Publish per-wave partial minima (lane l15==0 of each half holds rows p+8*half).
  if (l15 == 0) {
    #pragma unroll
    for (int p = 0; p < 8; ++p)
      partial[wave][p + 8 * half] = rmin[p];
  }
  __syncthreads();

  // Merge column-halves and finalize: one thread per row of the block's 64 rows.
  const int tid = threadIdx.x;
  if (tid < 64) {
    const int row = blockIdx.x * 64 + tid;
    if (row < N) {
      const int t = tid >> 4, r = tid & 15;
      float m = fminf(partial[t][r], partial[t + 4][r]);
      float d = sqrtf(fmaxf(m, 0.f));
      out[row] = fmaxf(d - R_SELF, 0.f);
    }
  }
}

extern "C" void kernel_launch(void* const* d_in, const int* in_sizes, int n_in,
                              void* d_out, int out_size, void* d_ws, size_t ws_size,
                              hipStream_t stream) {
  const float* x = (const float*)d_in[0];   // [N,128] fp32
  const float* s = (const float*)d_in[1];   // [M,128] fp32
  float* out = (float*)d_out;               // [N] fp32

  const int N = in_sizes[0] / DIMS;
  const int M = in_sizes[1] / DIMS;

  // Workspace layout (bf16 arrays then norms); ~12.6 MB for N=16384, M=8192.
  unsigned short* xh = (unsigned short*)d_ws;
  unsigned short* xl = xh + (size_t)N * DIMS;
  unsigned short* sh = xl + (size_t)N * DIMS;
  unsigned short* sl = sh + (size_t)M * DIMS;
  float* xn = (float*)(sl + (size_t)M * DIMS);
  float* sn = xn + N;

  hipLaunchKernelGGL(split_rows, dim3((N + 3) / 4), dim3(128), 0, stream, x, xh, xl, xn, N);
  hipLaunchKernelGGL(split_rows, dim3((M + 3) / 4), dim3(128), 0, stream, s, sh, sl, sn, M);

  const int blocks = (N + 63) / 64;         // 8 waves/block: 4 row tiles x 2 column halves
  hipLaunchKernelGGL(nsa_mindist_wmma, dim3(blocks), dim3(256), 0, stream,
                     xh, xl, xn, sh, sl, sn, out, N, M);
}
